// SparseCoreAttention_20229295964910
// MI455X (gfx1250) — matrix operand order repaired and checked by reading, not compiled, not run on hardware
//
#include <hip/hip_runtime.h>

typedef __attribute__((ext_vector_type(16))) _Float16 v16h;
typedef __attribute__((ext_vector_type(8)))  _Float16 v8h;
typedef __attribute__((ext_vector_type(8)))  float    v8f;

#define BB  16
#define SS  512
#define HH  16
#define HD  64
#define KBLK 32
#define NEGF (-3.0e38f)

union V16U { v16h v; v8h h[2]; };

__device__ __forceinline__ float rowmax16(float x) {
#pragma unroll
  for (int m = 8; m >= 1; m >>= 1) x = fmaxf(x, __shfl_xor(x, m, 16));
  return x;
}
__device__ __forceinline__ float rowsum16(float x) {
#pragma unroll
  for (int m = 8; m >= 1; m >>= 1) x += __shfl_xor(x, m, 16);
  return x;
}

__global__ __launch_bounds__(128)
void sparse_attn_flash_wmma(const float* __restrict__ qg,
                            const float* __restrict__ kg,
                            const float* __restrict__ vg,
                            const float* __restrict__ maskg,
                            float* __restrict__ out)
{
  // LDS: 8KB Q + 4KB K + 4KB V^T + 4KB P = 20KB (WGP has 320KB)
  __shared__ __align__(32) _Float16 sQ[64 * HD];        // (q_local, hd) row-major
  __shared__ __align__(32) _Float16 sK[KBLK * HD];      // (key_local, hd) row-major
  __shared__ __align__(32) _Float16 sV[HD * KBLK];      // transposed: (hd, key_local)
  __shared__ __align__(32) _Float16 sP[4][16 * KBLK];   // per-wave P bounce (q, key)

  const int tid  = threadIdx.x;
  const int wave = tid >> 5;
  const int lane = tid & 31;
  const int L    = lane & 15;   // N / column index inside 16-lane group
  const int hi   = lane >> 4;   // lane half

  const int b    = blockIdx.z;
  const int h    = blockIdx.y;
  const int qblk = blockIdx.x * 64;        // 64 query rows per WG
  const int qbase = qblk + wave * 16;      // 16 query rows per wave

  const float* maskb = maskg + (size_t)b * SS * SS;

  // ---- cooperative load + f32->f16 convert of the Q tile (64 x 64) ----
  for (int i = tid; i < 64 * (HD / 4); i += 128) {
    const int row = i >> 4;
    const int c4  = (i & 15) << 2;
    const float4 f = *(const float4*)(qg +
        (size_t)(((b * SS + qblk + row) * HH + h) * HD + c4));
    sQ[row * HD + c4 + 0] = (_Float16)f.x;
    sQ[row * HD + c4 + 1] = (_Float16)f.y;
    sQ[row * HD + c4 + 2] = (_Float16)f.z;
    sQ[row * HD + c4 + 3] = (_Float16)f.w;
  }
  __syncthreads();

  // ---- per-wave Q A-fragments (A 16x32 f16 layout, two HD chunks) ----
  // lane 0-15 : M=L, K = 0..15;  lane 16-31 : M=L, K = 16..31
  V16U aQ0, aQ1;
  {
    const _Float16* qr = &sQ[(wave * 16 + L) * HD];
    aQ0.h[0] = *(const v8h*)&qr[ 0 + hi * 16];
    aQ0.h[1] = *(const v8h*)&qr[ 8 + hi * 16];
    aQ1.h[0] = *(const v8h*)&qr[32 + hi * 16];
    aQ1.h[1] = *(const v8h*)&qr[40 + hi * 16];
  }

  const v8f vzero = {0.f, 0.f, 0.f, 0.f, 0.f, 0.f, 0.f, 0.f};
  v8f o[4];                 // O accumulator 16x64 f32, 4 N-chunks of 16
#pragma unroll
  for (int n = 0; n < 4; ++n) o[n] = vzero;
  float rmax[8], rsum[8];   // per-lane row stats; row m = r + hi*8
#pragma unroll
  for (int r = 0; r < 8; ++r) { rmax[r] = NEGF; rsum[r] = 0.f; }

  _Float16* sPw = sP[wave];

  for (int kb = 0; kb < SS; kb += KBLK) {
    __syncthreads();   // all waves done reading previous sK/sV
    // ---- cooperative load+convert K (row-major) and V (transposed) ----
    for (int i = tid; i < KBLK * (HD / 4); i += 128) {
      const int row = i >> 4;
      const int c4  = (i & 15) << 2;
      const size_t g = (size_t)(((b * SS + kb + row) * HH + h) * HD + c4);
      const float4 fk = *(const float4*)(kg + g);
      const float4 fv = *(const float4*)(vg + g);
      sK[row * HD + c4 + 0] = (_Float16)fk.x;
      sK[row * HD + c4 + 1] = (_Float16)fk.y;
      sK[row * HD + c4 + 2] = (_Float16)fk.z;
      sK[row * HD + c4 + 3] = (_Float16)fk.w;
      sV[(c4 + 0) * KBLK + row] = (_Float16)fv.x;
      sV[(c4 + 1) * KBLK + row] = (_Float16)fv.y;
      sV[(c4 + 2) * KBLK + row] = (_Float16)fv.z;
      sV[(c4 + 3) * KBLK + row] = (_Float16)fv.w;
      if (kb + KBLK < SS) {  // hint next block into cache (global_prefetch_b8)
        __builtin_prefetch(kg + g + (size_t)KBLK * HH * HD, 0, 1);
        __builtin_prefetch(vg + g + (size_t)KBLK * HH * HD, 0, 1);
      }
    }
    __syncthreads();

    // ---- scores: two 16x16 D tiles via chained 16x16x32 f16 WMMAs ----
    // B-frag (K^T 32x16): lane 0-15 K=0..15 contiguous, lane 16-31 K=16..31
    v8f s0 = vzero, s1 = vzero;
    {
      v16h bk;
      bk = *(const v16h*)&sK[L * HD + 0 + hi * 16];
      s0 = __builtin_amdgcn_wmma_f32_16x16x32_f16(false, aQ0.v, false, bk, (short)0, s0, false, false);
      bk = *(const v16h*)&sK[L * HD + 32 + hi * 16];
      s0 = __builtin_amdgcn_wmma_f32_16x16x32_f16(false, aQ1.v, false, bk, (short)0, s0, false, false);
      bk = *(const v16h*)&sK[(16 + L) * HD + 0 + hi * 16];
      s1 = __builtin_amdgcn_wmma_f32_16x16x32_f16(false, aQ0.v, false, bk, (short)0, s1, false, false);
      bk = *(const v16h*)&sK[(16 + L) * HD + 32 + hi * 16];
      s1 = __builtin_amdgcn_wmma_f32_16x16x32_f16(false, aQ1.v, false, bk, (short)0, s1, false, false);
    }

    // ---- masked online softmax in D layout (row m = r + hi*8) ----
#pragma unroll
    for (int r = 0; r < 8; ++r) {
      const int m  = r + hi * 8;
      const int qq = qbase + m;
      const float mk0 = maskb[qq * SS + kb + L];
      const float mk1 = maskb[qq * SS + kb + 16 + L];
      const float l0 = s0[r] * 0.125f;
      const float l1 = s1[r] * 0.125f;
      const float z0 = (mk0 > 0.f) ? l0 : NEGF;
      const float z1 = (mk1 > 0.f) ? l1 : NEGF;
      const float mx = rowmax16(fmaxf(z0, z1));
      const float nm = fmaxf(rmax[r], mx);
      const float corr = __expf(rmax[r] - nm);   // ==1 when both NEGF
      rmax[r] = nm;
      const float p0 = (mk0 > 0.f) ? __expf(l0 - nm) : 0.f;
      const float p1 = (mk1 > 0.f) ? __expf(l1 - nm) : 0.f;
      rsum[r] = rsum[r] * corr + rowsum16(p0 + p1);
      o[0][r] *= corr; o[1][r] *= corr; o[2][r] *= corr; o[3][r] *= corr;
      sPw[m * KBLK + L]      = (_Float16)p0;
      sPw[m * KBLK + 16 + L] = (_Float16)p1;
    }

    // ---- O += P(16x32) @ V(32x64): re-read P in A layout, V^T as B ----
    V16U aP;
    aP.h[0] = *(const v8h*)&sPw[L * KBLK + 0 + hi * 16];
    aP.h[1] = *(const v8h*)&sPw[L * KBLK + 8 + hi * 16];
#pragma unroll
    for (int n = 0; n < 4; ++n) {
      const v16h bv = *(const v16h*)&sV[(n * 16 + L) * KBLK + hi * 16];
      o[n] = __builtin_amdgcn_wmma_f32_16x16x32_f16(false, aP.v, false, bv, (short)0, o[n], false, false);
    }
  }

  // ---- normalize and store: out shape (S, B, H*HD) ----
#pragma unroll
  for (int r = 0; r < 8; ++r) {
    const int m  = r + hi * 8;
    const int qq = qbase + m;
    const float inv = 1.f / fmaxf(rsum[r], 1.1754944e-38f);
    float* op = out + ((size_t)qq * BB + b) * (HH * HD) + h * HD;
#pragma unroll
    for (int n = 0; n < 4; ++n)
      op[n * 16 + L] = o[n][r] * inv;
  }
}

extern "C" void kernel_launch(void* const* d_in, const int* in_sizes, int n_in,
                              void* d_out, int out_size, void* d_ws, size_t ws_size,
                              hipStream_t stream) {
  (void)in_sizes; (void)n_in; (void)out_size; (void)d_ws; (void)ws_size;
  const float* q = (const float*)d_in[0];
  const float* k = (const float*)d_in[1];
  const float* v = (const float*)d_in[2];
  const float* m = (const float*)d_in[3];
  dim3 grid(SS / 64, HH, BB);   // 8 x 16 x 16 = 2048 workgroups
  sparse_attn_flash_wmma<<<grid, dim3(128), 0, stream>>>(q, k, v, m, (float*)d_out);
}
